// KANLinear_8710193676795
// MI455X (gfx1250) — matrix operand order repaired and checked by reading, not compiled, not run on hardware
//
#include <hip/hip_runtime.h>
#include <hip/hip_bf16.h>

typedef __attribute__((ext_vector_type(16))) _Float16 v16h;
typedef __attribute__((ext_vector_type(8)))  _Float16 v8h;
typedef __attribute__((ext_vector_type(8)))  float    v8f;

#define IN_F   256
#define OUT_F  256
#define N_TOK  2048
#define KSPL   8                   // GRID + ORDER
#define H      0.4f                // (1 - -1)/GRID
#define QSTEP     (1.33f / 128.0f)
#define QINVSTEP  (128.0f / 1.33f)

// Cubic B-spline bases on uniform grid g[t] = (t-3)*H - 1, t=0..11 -> 8 bases
__device__ __forceinline__ void bspline8(float x, float* B) {
    float g[12];
#pragma unroll
    for (int t = 0; t < 12; ++t) g[t] = (float)(t - 3) * H - 1.0f;
    float b0[11];
#pragma unroll
    for (int j = 0; j < 11; ++j)
        b0[j] = (x >= g[j] && x < g[j + 1]) ? 1.0f : 0.0f;
    float b1[10];
#pragma unroll
    for (int j = 0; j < 10; ++j)
        b1[j] = ((x - g[j]) * b0[j] + (g[j + 2] - x) * b0[j + 1]) * (1.0f / (1.0f * H));
    float b2[9];
#pragma unroll
    for (int j = 0; j < 9; ++j)
        b2[j] = ((x - g[j]) * b1[j] + (g[j + 3] - x) * b1[j + 1]) * (1.0f / (2.0f * H));
#pragma unroll
    for (int j = 0; j < 8; ++j)
        B[j] = ((x - g[j]) * b2[j] + (g[j + 4] - x) * b2[j + 1]) * (1.0f / (3.0f * H));
}

// round-half-even + clamp to [-128,127]; input already scaled by 1/step
__device__ __forceinline__ float qclamp(float v) {
    float q = __builtin_rintf(v);                     // v_rndne_f32
    return fminf(fmaxf(q, -128.0f), 127.0f);          // v_med3_num_f32
}

// ---------------------------------------------------------------------------
// Pass A (64B records): per (n,i), the full per-lane WMMA A-fragment:
//   halfs  0..7  = [silu, bas0..bas6] (K0..7)    halfs  8..15 = [bas7, 0...] (K8..15)
//   halfs 16..31 = 0
// Lanes 0-15 load 32B at +0; lanes 16-31 load 32B at +32 -> loads fold
// directly into the WMMA source registers (no mov traffic).
// ---------------------------------------------------------------------------
__global__ __launch_bounds__(256)
void build_a64_kernel(const float* __restrict__ x, _Float16* __restrict__ aTab) {
    const size_t idx = (size_t)blockIdx.x * 256 + threadIdx.x;   // n*IN_F + i
    float xv = x[idx];
    float bas[8];
    bspline8(xv, bas);
    float sil = xv / (1.0f + __expf(-xv));

    v8h lo, hi, z;
    lo[0] = (_Float16)sil;
#pragma unroll
    for (int k = 0; k < 7; ++k) lo[k + 1] = (_Float16)bas[k];
#pragma unroll
    for (int e = 0; e < 8; ++e) { hi[e] = (_Float16)0.0f; z[e] = (_Float16)0.0f; }
    hi[0] = (_Float16)bas[7];

    _Float16* r = aTab + idx * 32;
    *(v8h*)(r)      = lo;
    *(v8h*)(r + 8)  = hi;
    *(v8h*)(r + 16) = z;
    *(v8h*)(r + 24) = z;
}

// ---------------------------------------------------------------------------
// Pass A (32B records) for the mid-size workspace path:
//   halfs 0..7 = [silu, bas0..bas6], halfs 8..15 = [bas7, 0...]
// ---------------------------------------------------------------------------
__global__ __launch_bounds__(256)
void build_a32_kernel(const float* __restrict__ x, _Float16* __restrict__ aTab) {
    const size_t idx = (size_t)blockIdx.x * 256 + threadIdx.x;
    float xv = x[idx];
    float bas[8];
    bspline8(xv, bas);
    float sil = xv / (1.0f + __expf(-xv));

    v8h lo, hi;
    lo[0] = (_Float16)sil;
#pragma unroll
    for (int k = 0; k < 7; ++k) lo[k + 1] = (_Float16)bas[k];
#pragma unroll
    for (int e = 0; e < 8; ++e) hi[e] = (_Float16)0.0f;
    hi[0] = (_Float16)bas[7];

    *(v8h*)(aTab + idx * 16)     = lo;
    *(v8h*)(aTab + idx * 16 + 8) = hi;
}

// ---------------------------------------------------------------------------
// Pass B: per (o,i) record of 16 halfs (32B), pre-scaled by 1/step so the
// WMMA output is acts/step directly. 8KB zero pad follows the table; lanes
// 16-31 stream through it so all lanes issue identical b128 loads.
// ---------------------------------------------------------------------------
__global__ __launch_bounds__(256)
void build_b_kernel(const float* __restrict__ bw, const float* __restrict__ sw,
                    const float* __restrict__ sc, const float* __restrict__ sel,
                    _Float16* __restrict__ bTab) {
    const size_t idx = (size_t)blockIdx.x * 256 + threadIdx.x;   // o*IN_F + i
    float se  = sel[idx] * QINVSTEP;
    float scv = sc[idx] * se;

    v8h lo, hi;
    lo[0] = (_Float16)(se * bw[idx]);
#pragma unroll
    for (int k = 0; k < 7; ++k) lo[k + 1] = (_Float16)(scv * sw[idx * KSPL + k]);
#pragma unroll
    for (int e = 0; e < 8; ++e) hi[e] = (_Float16)0.0f;
    hi[0] = (_Float16)(scv * sw[idx * KSPL + 7]);

    *(v8h*)(bTab + idx * 16)     = lo;
    *(v8h*)(bTab + idx * 16 + 8) = hi;
}

// Zero the 8KB pad after the B table (2048 dwords).
__global__ __launch_bounds__(256)
void zero_pad_kernel(unsigned int* __restrict__ p) {
    p[(size_t)blockIdx.x * 256 + threadIdx.x] = 0u;
}

// ---------------------------------------------------------------------------
// Main (64B A-records): fully divergence-free, both operands are full-width
// v16h loads that fold into WMMA source registers. Per i: 4x b128, 1 WMMA,
// 8x (rndne, med3) + 4x pk_add. acc holds exact integer levels; final
// fake_quant of the sum collapses to clamp(acc)*step exactly.
// ---------------------------------------------------------------------------
__global__ __launch_bounds__(256)
void kan_wmma_main64(const _Float16* __restrict__ aTab,
                     const _Float16* __restrict__ bTab,   // zero pad at end
                     float* __restrict__ out) {
    const int lane = threadIdx.x & 31;
    const int wave = threadIdx.x >> 5;
    const int wn   = wave >> 2;
    const int wo   = wave & 3;

    const int n0 = (blockIdx.x * 2 + wn) * 16;
    const int o0 = (blockIdx.y * 4 + wo) * 16;

    const int  m  = lane & 15;
    const bool hi = lane >= 16;

    const _Float16* aP = aTab + ((size_t)(n0 + m) * IN_F) * 32 + (hi ? 16 : 0);
    const _Float16* bP = bTab + (hi ? (size_t)OUT_F * IN_F * 16
                                    : (size_t)(o0 + m) * IN_F * 16);

    v8f acc = {};

#pragma unroll 2
    for (int i = 0; i < IN_F; ++i) {
        v16h a = *(const v16h*)(aP + (size_t)i * 32);    // 2x global_load_b128
        v16h b = *(const v16h*)(bP + (size_t)i * 16);    // 2x global_load_b128

        v8f c = {};
        c = __builtin_amdgcn_wmma_f32_16x16x32_f16(
                false, a, false, b, (short)0, c, false, false);
#pragma unroll
        for (int r = 0; r < 8; ++r) acc[r] += qclamp(c[r]);
    }

#pragma unroll
    for (int r = 0; r < 8; ++r) {
        int M = r + (hi ? 8 : 0);
        out[(size_t)(n0 + M) * OUT_F + (o0 + m)] =
            fminf(fmaxf(acc[r], -128.0f), 127.0f) * QSTEP;
    }
}

// ---------------------------------------------------------------------------
// Main (32B A-records): mid-size workspace path (A upper half zeroed in-reg).
// ---------------------------------------------------------------------------
__global__ __launch_bounds__(256)
void kan_wmma_main32(const _Float16* __restrict__ aTab,
                     const _Float16* __restrict__ bTab,
                     float* __restrict__ out) {
    const int lane = threadIdx.x & 31;
    const int wave = threadIdx.x >> 5;
    const int wn   = wave >> 2;
    const int wo   = wave & 3;

    const int n0 = (blockIdx.x * 2 + wn) * 16;
    const int o0 = (blockIdx.y * 4 + wo) * 16;

    const int  m  = lane & 15;
    const bool hi = lane >= 16;

    const _Float16* aP = aTab + ((size_t)(n0 + m) * IN_F) * 16;
    const _Float16* bP = bTab + (hi ? (size_t)OUT_F * IN_F * 16
                                    : (size_t)(o0 + m) * IN_F * 16);

    v8f acc = {};

#pragma unroll 2
    for (int i = 0; i < IN_F; ++i) {
        v16h a = *(const v16h*)(aP + (size_t)i * 16);
        if (hi) {
#pragma unroll
            for (int e = 0; e < 16; ++e) a[e] = (_Float16)0.0f;
        }

        v16h b = *(const v16h*)(bP + (size_t)i * 16);

        v8f c = {};
        c = __builtin_amdgcn_wmma_f32_16x16x32_f16(
                false, a, false, b, (short)0, c, false, false);
#pragma unroll
        for (int r = 0; r < 8; ++r) acc[r] += qclamp(c[r]);
    }

#pragma unroll
    for (int r = 0; r < 8; ++r) {
        int M = r + (hi ? 8 : 0);
        out[(size_t)(n0 + M) * OUT_F + (o0 + m)] =
            fminf(fmaxf(acc[r], -128.0f), 127.0f) * QSTEP;
    }
}

// ---------------------------------------------------------------------------
// Fallback (insufficient workspace): fused single pass, same math.
// ---------------------------------------------------------------------------
__global__ __launch_bounds__(256)
void kan_wmma_fused(const float* __restrict__ x,  const float* __restrict__ bw,
                    const float* __restrict__ sw, const float* __restrict__ sc,
                    const float* __restrict__ sel, float* __restrict__ out) {
    const int lane = threadIdx.x & 31;
    const int wave = threadIdx.x >> 5;
    const int wn   = wave >> 2;
    const int wo   = wave & 3;
    const int n0 = (blockIdx.x * 2 + wn) * 16;
    const int o0 = (blockIdx.y * 4 + wo) * 16;
    const int  m  = lane & 15;
    const bool hi = lane >= 16;
    const int n = n0 + m;
    const int o = o0 + m;

    v8f acc = {};
    for (int i = 0; i < IN_F; ++i) {
        float xv = x[(size_t)n * IN_F + i];
        float bas[8];
        bspline8(xv, bas);
        float sil = xv / (1.0f + __expf(-xv));

        v16h a;
#pragma unroll
        for (int e = 0; e < 16; ++e) a[e] = (_Float16)0.0f;
        if (!hi) {
            a[0] = (_Float16)sil;
#pragma unroll
            for (int k = 0; k < 8; ++k) a[k + 1] = (_Float16)bas[k];
        }

        v16h b;
#pragma unroll
        for (int e = 0; e < 16; ++e) b[e] = (_Float16)0.0f;
        if (!hi) {
            const size_t oi = (size_t)o * IN_F + i;
            float se  = sel[oi] * QINVSTEP;
            float scv = sc[oi] * se;
            b[0] = (_Float16)(se * bw[oi]);
#pragma unroll
            for (int k = 0; k < 8; ++k)
                b[k + 1] = (_Float16)(scv * sw[oi * KSPL + k]);
        }

        v8f c = {};
        c = __builtin_amdgcn_wmma_f32_16x16x32_f16(
                false, a, false, b, (short)0, c, false, false);
#pragma unroll
        for (int r = 0; r < 8; ++r) acc[r] += qclamp(c[r]);
    }
#pragma unroll
    for (int r = 0; r < 8; ++r) {
        int M = r + (hi ? 8 : 0);
        out[(size_t)(n0 + M) * OUT_F + (o0 + m)] =
            fminf(fmaxf(acc[r], -128.0f), 127.0f) * QSTEP;
    }
}

extern "C" void kernel_launch(void* const* d_in, const int* in_sizes, int n_in,
                              void* d_out, int out_size, void* d_ws, size_t ws_size,
                              hipStream_t stream) {
    const float* x   = (const float*)d_in[0];   // (2048, 256)
    const float* bw  = (const float*)d_in[1];   // (256, 256)
    const float* sw  = (const float*)d_in[2];   // (256, 256, 8)
    const float* sc  = (const float*)d_in[3];   // (256, 256)
    const float* sel = (const float*)d_in[4];   // (256, 256)
    float* out = (float*)d_out;                 // (2048, 256)

    const size_t a64Bytes = (size_t)N_TOK * IN_F * 32 * sizeof(_Float16); // 32 MB
    const size_t a32Bytes = (size_t)N_TOK * IN_F * 16 * sizeof(_Float16); // 16 MB
    const size_t bBytes   = (size_t)OUT_F * IN_F * 16 * sizeof(_Float16); //  2 MB
    const size_t padBytes = (size_t)IN_F * 16 * sizeof(_Float16);         //  8 KB

    dim3 grid(N_TOK / 32, OUT_F / 64);          // 64 x 4
    dim3 block(256);                            // 8 wave32 waves

    if (ws_size >= a64Bytes + bBytes + padBytes) {
        _Float16* aTab = (_Float16*)d_ws;
        _Float16* bTab = (_Float16*)((char*)d_ws + a64Bytes);
        unsigned int* pad = (unsigned int*)((char*)d_ws + a64Bytes + bBytes);
        zero_pad_kernel<<<(int)(padBytes / 4 / 256), 256, 0, stream>>>(pad);
        build_a64_kernel<<<(N_TOK * IN_F) / 256, 256, 0, stream>>>(x, aTab);
        build_b_kernel<<<(OUT_F * IN_F) / 256, 256, 0, stream>>>(bw, sw, sc, sel, bTab);
        kan_wmma_main64<<<grid, block, 0, stream>>>(aTab, bTab, out);
    } else if (ws_size >= a32Bytes + bBytes + padBytes) {
        _Float16* aTab = (_Float16*)d_ws;
        _Float16* bTab = (_Float16*)((char*)d_ws + a32Bytes);
        unsigned int* pad = (unsigned int*)((char*)d_ws + a32Bytes + bBytes);
        zero_pad_kernel<<<(int)(padBytes / 4 / 256), 256, 0, stream>>>(pad);
        build_a32_kernel<<<(N_TOK * IN_F) / 256, 256, 0, stream>>>(x, aTab);
        build_b_kernel<<<(OUT_F * IN_F) / 256, 256, 0, stream>>>(bw, sw, sc, sel, bTab);
        kan_wmma_main32<<<grid, block, 0, stream>>>(aTab, bTab, out);
    } else {
        kan_wmma_fused<<<grid, block, 0, stream>>>(x, bw, sw, sc, sel, out);
    }
}
